// RefEncoder_49727131353592
// MI455X (gfx1250) — matrix operand order repaired and checked by reading, not compiled, not run on hardware
//
#include <hip/hip_runtime.h>
#include <cstdint>
#include <cstddef>

typedef __attribute__((ext_vector_type(16))) _Float16 v16h;
typedef __attribute__((ext_vector_type(8)))  float    v8f;

__device__ __forceinline__ float gelu_f(float x) {
  const float c0 = 0.7978845608028654f;   // sqrt(2/pi)
  const float c1 = 0.044715f;
  float x3 = x * x * x;
  return 0.5f * x * (1.0f + tanhf(c0 * (x + c1 * x3)));
}

// --------------------------------------------------------------------------
// TDM: async 2D tile load (element size 2B) Global -> LDS via Tensor Data
// Mover. Descriptor bit layout per CDNA5 ISA ch.10 (D# groups 0/1).
// --------------------------------------------------------------------------
#if defined(__has_builtin)
#if __has_builtin(__builtin_amdgcn_tensor_load_to_lds)
#define HAVE_TDM 1
#endif
#endif
#ifndef HAVE_TDM
#define HAVE_TDM 0
#endif

__device__ __forceinline__ void tdm_load_2d_f16(const void* gtile,
                                                uint32_t lds_byte_addr,
                                                uint32_t tensor_w_elems,
                                                uint32_t tensor_h_rows,
                                                uint32_t row_stride_elems,
                                                uint32_t tile_w_elems,
                                                uint32_t tile_h_rows) {
#if HAVE_TDM
  typedef __attribute__((ext_vector_type(4))) unsigned int u32x4;
  typedef __attribute__((ext_vector_type(4))) int i32x4;
  typedef __attribute__((ext_vector_type(8))) int i32x8;
  uint64_t ga = (uint64_t)(uintptr_t)gtile;
  u32x4 g0;
  g0[0] = 1u;                                        // count=1 (valid), user D#
  g0[1] = lds_byte_addr;                             // lds_addr
  g0[2] = (uint32_t)(ga & 0xFFFFFFFFu);              // global_addr[31:0]
  g0[3] = (uint32_t)((ga >> 32) & 0x1FFFFFFu) | (2u << 30);  // [56:32] | type=2
  i32x8 g1;
  g1[0] = (int)(1u << 16);                           // data_size=1 -> 2 bytes
  g1[1] = (int)((tensor_w_elems & 0xFFFFu) << 16);   // tensor_dim0[15:0]
  g1[2] = (int)((tensor_w_elems >> 16) & 0xFFFFu) |
          (int)((tensor_h_rows & 0xFFFFu) << 16);    // dim0[31:16] | dim1[15:0]
  g1[3] = (int)((tensor_h_rows >> 16) & 0xFFFFu) |
          (int)(tile_w_elems << 16);                 // dim1[31:16] | tile_dim0
  g1[4] = (int)tile_h_rows;                          // tile_dim1 (tile_dim2=0)
  g1[5] = (int)row_stride_elems;                     // tensor_dim0_stride[31:0]
  g1[6] = 0;                                         // stride[47:32] | d1s[15:0]
  g1[7] = 0;                                         // d1s[47:16]
  i32x4 z4 = {0, 0, 0, 0};
#if __clang_major__ >= 23
  i32x8 z8 = {0, 0, 0, 0, 0, 0, 0, 0};
  __builtin_amdgcn_tensor_load_to_lds(g0, g1, z4, z4, z8, 0);
#else
  __builtin_amdgcn_tensor_load_to_lds(g0, g1, z4, z4, 0);
#endif
  __builtin_amdgcn_s_wait_tensorcnt(0);
#else
  (void)gtile; (void)lds_byte_addr; (void)tensor_w_elems; (void)tensor_h_rows;
  (void)row_stride_elems; (void)tile_w_elems; (void)tile_h_rows;
#endif
}

// --------------------------------------------------------------------------
// Prep: fp32 NCHW [16,3,256,256] -> f16 NHWC [16,256,256,4] (ch3 = 0), norm.
// --------------------------------------------------------------------------
__global__ void prep_input_kernel(const float* __restrict__ x,
                                  _Float16* __restrict__ out) {
  int i = blockIdx.x * blockDim.x + threadIdx.x;      // 16*256*256*4
  if (i >= 16 * 256 * 256 * 4) return;
  int c = i & 3;
  int w = (i >> 2) & 255;
  int h = (i >> 10) & 255;
  int n = i >> 18;
  float v = 0.f;
  if (c < 3) v = (x[(((size_t)n * 3 + c) * 256 + h) * 256 + w] - 0.5f) * 2.0f;
  out[i] = (_Float16)v;
}

// --------------------------------------------------------------------------
// Prep: fp32 OIHW weights -> f16 [KPAD][CO], K = (r*3+s)*CIP + ci, zero-pad.
// --------------------------------------------------------------------------
template <int CIO, int CIP, int CO, int KPAD>
__global__ void prep_weights_kernel(const float* __restrict__ w,
                                    _Float16* __restrict__ out) {
  int i = blockIdx.x * blockDim.x + threadIdx.x;      // KPAD*CO
  if (i >= KPAD * CO) return;
  int k  = i / CO, co = i % CO;
  int rs = k / CIP, ci = k % CIP;
  float v = 0.f;
  if (rs < 9 && ci < CIO) v = w[((size_t)co * CIO + ci) * 9 + rs];
  out[i] = (_Float16)v;
}

// --------------------------------------------------------------------------
// Strided (2x) 3x3 conv as implicit GEMM with v_wmma_f32_16x16x32_f16.
// Block: 128 threads (4 waves). Tile: 16 output pixels x 64 output channels.
// A (16 x 32 im2col) staged by VALU; B (32 x 64 weights) staged by the
// Tensor Data Mover (tensor_load_to_lds + s_wait_tensorcnt).
// --------------------------------------------------------------------------
template <int CI, int CO, int HI, int WI, int HO, int WO, bool GELU>
__global__ __launch_bounds__(128)
void conv_s2_wmma_kernel(const _Float16* __restrict__ in,
                         const _Float16* __restrict__ wht,
                         const float* __restrict__ bias,
                         _Float16* __restrict__ out) {
  constexpr int KPAD = (CI * 9 + 31) & ~31;

  __shared__ __align__(16) _Float16 sA[16 * 32];
  __shared__ __align__(16) _Float16 sB[32 * 64];

  const int tid  = threadIdx.x;
  const int wave = tid >> 5;
  const int lane = tid & 31;
  const int pixbase = blockIdx.x * 16;
  const int cobase  = blockIdx.y * 64;

  // A-staging coords: thread -> (pixel, 4-wide K chunk)
  const int sp  = tid >> 3;             // 0..15
  const int sk4 = (tid & 7) * 4;        // 0,4,...,28
  const int pixS = pixbase + sp;
  const int nS  = pixS / (HO * WO);
  const int ohS = (pixS / WO) % HO;
  const int owS = pixS % WO;
  // B fallback staging coords
  const int brow = tid >> 2;            // 0..31
  const int bcg  = (tid & 3) * 16;      // 0,16,32,48

  v8f acc = {};

  for (int k0 = 0; k0 < KPAD; k0 += 32) {
    // ---- stage A tile (im2col), 4 contiguous channels per thread ----
    {
      int K  = k0 + sk4;
      int rs = K / CI;
      int ci = K % CI;
      uint2 val; val.x = 0u; val.y = 0u;
      if (rs < 9) {
        int r = rs / 3, s = rs % 3;
        int ih = ohS * 2 + r;           // SAME pad: pad_lo = 0, pad_hi = 1
        int iw = owS * 2 + s;
        if (ih < HI && iw < WI) {
          val = *reinterpret_cast<const uint2*>(
              in + ((size_t)(nS * HI + ih) * WI + iw) * CI + ci);
        }
      }
      *reinterpret_cast<uint2*>(&sA[sp * 32 + sk4]) = val;
    }
    // ---- stage B tile (weights 32 x 64) via TDM (wave 0 issues) ----
#if HAVE_TDM
    if (wave == 0) {
      tdm_load_2d_f16(wht + (size_t)k0 * CO + cobase,
                      (uint32_t)(uintptr_t)(&sB[0]),
                      /*tensor_w*/ CO, /*tensor_h*/ KPAD,
                      /*row_stride*/ CO, /*tile_w*/ 64, /*tile_h*/ 32);
    }
#else
    {
      const _Float16* src = wht + (size_t)(k0 + brow) * CO + cobase + bcg;
      uint4 v0 = *reinterpret_cast<const uint4*>(src);
      uint4 v1 = *reinterpret_cast<const uint4*>(src + 8);
      *reinterpret_cast<uint4*>(&sB[brow * 64 + bcg])     = v0;
      *reinterpret_cast<uint4*>(&sB[brow * 64 + bcg + 8]) = v1;
    }
#endif
    __syncthreads();

    if (k0 + 32 < KPAD)  // hint next weight tile toward L2 (global_prefetch_b8)
      __builtin_prefetch(wht + (size_t)(k0 + 32) * CO + cobase, 0, 3);

    // ---- build A fragment (16x32 f16, ISA lane layout) ----
    v16h af = {};
    {
      int row = lane & 15;
      int kb  = (lane >> 4) * 8;        // lanes 16-31 hold K+8 / K+24 halves
      uint4* aq = reinterpret_cast<uint4*>(&af);
      aq[0] = *reinterpret_cast<const uint4*>(&sA[row * 32 + kb]);       // K kb..kb+7
      aq[1] = *reinterpret_cast<const uint4*>(&sA[row * 32 + 16 + kb]);  // K 16+kb..
    }
    // ---- build B fragment (32x16 f16): lane<16 -> K0..7,16..23, lane>=16 -> K8..15,24..31
    v16h bf = {};
    {
      int col = (wave << 4) + (lane & 15);
      int kb  = (lane >> 4) * 8;
#pragma unroll
      for (int j = 0; j < 16; ++j) bf[j] = sB[(kb + j + (j & 8)) * 64 + col];
    }

    acc = __builtin_amdgcn_wmma_f32_16x16x32_f16(
        false, af, false, bf, (short)0, acc, false, false);
    __syncthreads();
  }

  // ---- epilogue: bias (+GELU), store f16 NHWC ----
  const int co = cobase + (wave << 4) + (lane & 15);
  const float bv = bias[co];
#pragma unroll
  for (int r = 0; r < 8; ++r) {
    int row = ((lane >> 4) << 3) + r;   // D-matrix row layout
    int pix = pixbase + row;
    float v = acc[r] + bv;
    if (GELU) v = gelu_f(v);
    out[(size_t)pix * CO + co] = (_Float16)v;
  }
}

// --------------------------------------------------------------------------
// G[m] = Wk^T * Wq : G[m][dp][d] = sum_c wk[m][c][dp]*wq[m][c][d]
// --------------------------------------------------------------------------
__global__ void g_kernel(const float* __restrict__ wq, const float* __restrict__ wk,
                         float* __restrict__ G) {
  int i = blockIdx.x * blockDim.x + threadIdx.x;      // 4*64*64
  if (i >= 4 * 64 * 64) return;
  int m = i >> 12, dp = (i >> 6) & 63, d = i & 63;
  float s = 0.f;
  for (int c = 0; c < 64; ++c)
    s += wk[((size_t)m * 64 + c) * 64 + dp] * wq[((size_t)m * 64 + c) * 64 + d];
  G[i] = s;
}

// --------------------------------------------------------------------------
// E[m] = codebook[m] * G[m]  -> f16 [4][256][64]
// --------------------------------------------------------------------------
__global__ void e_kernel(const float* __restrict__ cb, const float* __restrict__ G,
                         _Float16* __restrict__ E) {
  int i = blockIdx.x * blockDim.x + threadIdx.x;      // 4*256*64
  if (i >= 4 * 256 * 64) return;
  int m = i >> 14, k = (i >> 6) & 255, d = i & 63;
  float s = 0.f;
  for (int dp = 0; dp < 64; ++dp)
    s += cb[((size_t)m * 256 + k) * 64 + dp] * G[((size_t)m * 64 + dp) * 64 + d];
  E[i] = (_Float16)s;
}

// --------------------------------------------------------------------------
// scores[pix][m][k] = sum_d E[m][k][d] * lat[pix][m*64+d]; fused argmax over k.
// WMMA tile: M=16 codes x N=16 pixels, K=64 (2 k-steps). Branchless argmax.
// --------------------------------------------------------------------------
__global__ __launch_bounds__(128)
void scores_argmax_kernel(const _Float16* __restrict__ lat,
                          const _Float16* __restrict__ E,
                          float* __restrict__ out) {
  __shared__ float sval[4][16];
  __shared__ int   sidx[4][16];

  const int tid  = threadIdx.x;
  const int wave = tid >> 5;
  const int lane = tid & 31;
  const int pixbase = blockIdx.x * 16;
  const int m = blockIdx.y;
  const int pix = pixbase + (lane & 15);

  const _Float16* latp = lat + (size_t)pix * 256 + m * 64;
  const _Float16* Em   = E + (size_t)m * 256 * 64;

  float best = -3.4e38f;
  int   besti = 0;

#pragma unroll
  for (int ct = 0; ct < 4; ++ct) {
    const int codebase = (wave * 4 + ct) * 16;
    v8f acc = {};
#pragma unroll
    for (int ks = 0; ks < 2; ++ks) {
      const int d0 = ks * 32;
      v16h af = {};
      {
        int code = codebase + (lane & 15);
        int kb   = (lane >> 4) * 8;
        const _Float16* ep = Em + (size_t)code * 64 + d0 + kb;
        uint4* aq = reinterpret_cast<uint4*>(&af);
        aq[0] = *reinterpret_cast<const uint4*>(ep);
        aq[1] = *reinterpret_cast<const uint4*>(ep + 16);
      }
      v16h bf = {};
      {
        const _Float16* bp = latp + d0 + ((lane >> 4) << 3);
        uint4* bq = reinterpret_cast<uint4*>(&bf);
        bq[0] = *reinterpret_cast<const uint4*>(bp);
        bq[1] = *reinterpret_cast<const uint4*>(bp + 16);
      }
      acc = __builtin_amdgcn_wmma_f32_16x16x32_f16(
          false, af, false, bf, (short)0, acc, false, false);
    }
    // codes visited in ascending order within a lane -> strict '>' keeps first
#pragma unroll
    for (int r = 0; r < 8; ++r) {
      int code = codebase + ((lane >> 4) << 3) + r;
      float v = acc[r];
      bool t = v > best;
      best  = t ? v : best;
      besti = t ? code : besti;
    }
  }
  // lanes L and L+16 hold the same pixel, different code rows (branchless)
  {
    float ov = __shfl_xor(best, 16, 32);
    int   oi = __shfl_xor(besti, 16, 32);
    bool t = (ov > best) || (ov == best && oi < besti);
    best  = t ? ov : best;
    besti = t ? oi : besti;
  }

  if (lane < 16) { sval[wave][lane] = best; sidx[wave][lane] = besti; }
  __syncthreads();
  if (tid < 16) {
    float bv = sval[0][tid]; int bi = sidx[0][tid];
#pragma unroll
    for (int w2 = 1; w2 < 4; ++w2) {
      float v = sval[w2][tid]; int i2 = sidx[w2][tid];
      bool t = (v > bv) || (v == bv && i2 < bi);
      bv = t ? v : bv;
      bi = t ? i2 : bi;
    }
    out[(size_t)(pixbase + tid) * 4 + m] = (float)bi;
  }
}

__global__ void tail_kernel(float* __restrict__ out) {
  if (threadIdx.x == 0) { out[16384] = 256.0f; out[16385] = 256.0f; }
}

// --------------------------------------------------------------------------
// Launch
// --------------------------------------------------------------------------
extern "C" void kernel_launch(void* const* d_in, const int* in_sizes, int n_in,
                              void* d_out, int out_size, void* d_ws, size_t ws_size,
                              hipStream_t stream) {
  (void)in_sizes; (void)n_in; (void)out_size; (void)ws_size;

  const float* x  = (const float*)d_in[0];
  const float* w1 = (const float*)d_in[1];
  const float* b1 = (const float*)d_in[2];
  const float* w2 = (const float*)d_in[3];
  const float* b2 = (const float*)d_in[4];
  const float* w3 = (const float*)d_in[5];
  const float* b3 = (const float*)d_in[6];
  const float* w4 = (const float*)d_in[7];
  const float* b4 = (const float*)d_in[8];
  const float* cb = (const float*)d_in[9];
  const float* wq = (const float*)d_in[10];
  const float* wk = (const float*)d_in[11];
  float* out = (float*)d_out;

  // Workspace layout (bytes). bufA holds a0 -> a2 -> a4; bufB holds a1 -> a3.
  char* ws = (char*)d_ws;
  const size_t SZ_BUFA = (size_t)16 * 64 * 64 * 128 * sizeof(_Float16);  // 16.8 MB
  const size_t SZ_BUFB = (size_t)16 * 128 * 128 * 64 * sizeof(_Float16); // 33.6 MB
  _Float16* bufA = (_Float16*)(ws);
  _Float16* bufB = (_Float16*)(ws + SZ_BUFA);
  char* p = ws + SZ_BUFA + SZ_BUFB;
  _Float16* w1p = (_Float16*)p; p += (size_t)64   * 64  * sizeof(_Float16);
  _Float16* w2p = (_Float16*)p; p += (size_t)576  * 128 * sizeof(_Float16);
  _Float16* w3p = (_Float16*)p; p += (size_t)1152 * 256 * sizeof(_Float16);
  _Float16* w4p = (_Float16*)p; p += (size_t)2304 * 256 * sizeof(_Float16);
  float*    Gm  = (float*)p;    p += (size_t)4 * 64 * 64 * sizeof(float);
  _Float16* Em  = (_Float16*)p;

  _Float16* a0 = bufA;   // [16,256,256,4]
  _Float16* a1 = bufB;   // [16,128,128,64]
  _Float16* a2 = bufA;   // [16,64,64,128]
  _Float16* a3 = bufB;   // [16,32,32,256]
  _Float16* a4 = bufA;   // [16,16,16,256] latent

  // 1) input + weight prep
  prep_input_kernel<<<(16 * 256 * 256 * 4 + 255) / 256, 256, 0, stream>>>(x, a0);
  prep_weights_kernel<3,   4,   64,   64><<<( 64   * 64  + 255) / 256, 256, 0, stream>>>(w1, w1p);
  prep_weights_kernel<64,  64,  128, 576><<<(576  * 128 + 255) / 256, 256, 0, stream>>>(w2, w2p);
  prep_weights_kernel<128, 128, 256, 1152><<<(1152 * 256 + 255) / 256, 256, 0, stream>>>(w3, w3p);
  prep_weights_kernel<256, 256, 256, 2304><<<(2304 * 256 + 255) / 256, 256, 0, stream>>>(w4, w4p);

  // 2) encoder: implicit-GEMM convs on WMMA (weights staged by TDM)
  conv_s2_wmma_kernel<4,   64,  256, 256, 128, 128, true >
      <<<dim3(16 * 128 * 128 / 16, 1), 128, 0, stream>>>(a0, w1p, b1, a1);
  conv_s2_wmma_kernel<64,  128, 128, 128, 64,  64,  true >
      <<<dim3(16 * 64 * 64 / 16, 2), 128, 0, stream>>>(a1, w2p, b2, a2);
  conv_s2_wmma_kernel<128, 256, 64,  64,  32,  32,  true >
      <<<dim3(16 * 32 * 32 / 16, 4), 128, 0, stream>>>(a2, w3p, b3, a3);
  conv_s2_wmma_kernel<256, 256, 32,  32,  16,  16,  false>
      <<<dim3(16 * 16 * 16 / 16, 4), 128, 0, stream>>>(a3, w4p, b4, a4);

  // 3) fused codebook tail: G = Wk^T Wq ; E = cb*G ; scores+argmax via WMMA
  g_kernel<<<(4 * 64 * 64 + 255) / 256, 256, 0, stream>>>(wq, wk, Gm);
  e_kernel<<<(4 * 256 * 64 + 255) / 256, 256, 0, stream>>>(cb, Gm, Em);
  scores_argmax_kernel<<<dim3(256, 4), 128, 0, stream>>>(a4, Em, out);

  tail_kernel<<<1, 32, 0, stream>>>(out);
}
